// Network_49203145343095
// MI455X (gfx1250) — matrix operand order repaired and checked by reading, not compiled, not run on hardware
//
#include <hip/hip_runtime.h>
#include <hip/hip_bf16.h>

// ---------------------------------------------------------------------------
// Shapes: B=64, L=512, D=768, C=32, H200=200, H=100
// Outputs (flat, in order): pred_pol 64*32*3 | pred_cause 64*32*2 |
//   sq_out 64*768 | ds1_out 64*32*768 | fsq_out 64*768   (total 1,681,408 f32)
// ---------------------------------------------------------------------------

typedef __attribute__((ext_vector_type(16))) __bf16 v16bf;
typedef __attribute__((ext_vector_type(8)))  float  v8f;

// ---------------------------------------------------------------------------
// Generic bf16 WMMA GEMM:  C[m,n] = sum_k A'[m,k]*B'[k,n] (+bias1[n]+bias2[n])
//   A row m -> physical row (aGroup>0 ? m + m/aGroup + 1 : m), addr A[row*lda+k]
//   B: transB ? B[n*ldb+k] : B[k*ldb+n]
// One wave per 16x16 C tile, v_wmma_f32_16x16x32_bf16 over K.
// Fragment layouts per CDNA5 ISA 7.12.2:
//   A (16x32 bf16): lane l elem e -> m=l%16, k = (e<8?e:e+8) + (l/16)*8
//     (even/odd element pairs are k-contiguous -> float2 loads on fast path)
//   B (32x16 bf16): lane l elem e -> n=l%16, k = (e<8?e:e+8) + (l/16)*8
//   C/D (v8f):      lane l reg  r -> n=l%16, m = r + (l/16)*8
// Interior tiles take an unmasked fast path; edge tiles take the masked path
// (wave-uniform branch, EXEC stays all-ones across every WMMA).
// ---------------------------------------------------------------------------
__global__ __launch_bounds__(256) void gemm_bf16_wmma(
    const float* __restrict__ A, const float* __restrict__ Bm,
    float* __restrict__ C, const float* __restrict__ bias1,
    const float* __restrict__ bias2,
    int M, int N, int Kd, int lda, int ldb, int ldc,
    int tilesN, int totalTiles, int aGroup, int transB)
{
  int wave = threadIdx.x >> 5;
  int lane = threadIdx.x & 31;
  int tile = blockIdx.x * 8 + wave;
  if (tile >= totalTiles) return;           // wave-uniform exit
  int nT = tile % tilesN, mT = tile / tilesN;
  int hlf = lane >> 4, l16 = lane & 15;
  int n0 = nT * 16;

  v8f acc = {};
  bool interior = (mT * 16 + 16 <= M) && (n0 + 16 <= N);   // wave-uniform
  int kMain = Kd & ~31;

  if (interior) {
    int  m    = mT * 16 + l16;
    long aRow = (long)(aGroup > 0 ? (m + m / aGroup + 1) : m) * (long)lda;
    const float* aP = A + aRow;
    // ---- unmasked main K loop ----
    for (int k0 = 0; k0 < kMain; k0 += 32) {
      v16bf a, b;
#pragma unroll
      for (int r = 0; r < 8; ++r) {          // pairs (2r,2r+1): contiguous k
        int k = k0 + (r < 4 ? 2 * r : 16 + 2 * (r - 4)) + hlf * 8;
        float2 av = *(const float2*)(aP + k);
        a[2 * r]     = (__bf16)av.x;
        a[2 * r + 1] = (__bf16)av.y;
      }
      int nb = n0 + l16;
#pragma unroll
      for (int e = 0; e < 16; ++e) {
        int kb = k0 + (e < 8 ? e : e + 8) + hlf * 8;
        b[e] = (__bf16)(transB ? Bm[(long)nb * ldb + kb] : Bm[(long)kb * ldb + nb]);
      }
      acc = __builtin_amdgcn_wmma_f32_16x16x32_bf16(
          false, a, false, b, (short)0, acc, false, false);
    }
    // ---- masked K tail (k only) ----
    if (kMain < Kd) {
      v16bf a, b;
#pragma unroll
      for (int e = 0; e < 16; ++e) {
        int  k   = kMain + (e < 8 ? e : e + 8) + hlf * 8;
        bool kOk = k < Kd;
        float av = aP[kOk ? k : (Kd - 1)];
        a[e] = (__bf16)(kOk ? av : 0.0f);
      }
      int nb = n0 + l16;
#pragma unroll
      for (int e = 0; e < 16; ++e) {
        int  kb   = kMain + (e < 8 ? e : e + 8) + hlf * 8;
        bool kbOk = kb < Kd;
        int  kbc  = kbOk ? kb : (Kd - 1);
        float bv = transB ? Bm[(long)nb * ldb + kbc]
                          : Bm[(long)kbc * ldb + nb];
        b[e] = (__bf16)(kbOk ? bv : 0.0f);
      }
      acc = __builtin_amdgcn_wmma_f32_16x16x32_bf16(
          false, a, false, b, (short)0, acc, false, false);
    }
    // ---- store (no bounds checks), bias hoisted to one load per lane ----
    int   n    = n0 + l16;
    float badd = (bias1 ? bias1[n] : 0.0f) + (bias2 ? bias2[n] : 0.0f);
#pragma unroll
    for (int r = 0; r < 8; ++r) {
      int mm = mT * 16 + r + hlf * 8;
      C[(long)mm * ldc + n] = acc[r] + badd;
    }
  } else {
    // ---- fully masked edge path ----
    int  m   = mT * 16 + l16;
    bool mOk = m < M;
    int  mc  = mOk ? m : (M - 1);
    long aRow = (long)(aGroup > 0 ? (mc + mc / aGroup + 1) : mc) * (long)lda;
    for (int k0 = 0; k0 < Kd; k0 += 32) {
      v16bf a, b;
#pragma unroll
      for (int e = 0; e < 16; ++e) {
        int  k   = k0 + (e < 8 ? e : e + 8) + hlf * 8;
        bool kOk = k < Kd;
        float av = A[aRow + (kOk ? k : (Kd - 1))];
        a[e] = (__bf16)((mOk && kOk) ? av : 0.0f);
      }
      int  n   = n0 + l16;
      bool nOk = n < N;
      int  nc  = nOk ? n : (N - 1);
#pragma unroll
      for (int e = 0; e < 16; ++e) {
        int  kb   = k0 + (e < 8 ? e : e + 8) + hlf * 8;
        bool kbOk = kb < Kd;
        int  kbc  = kbOk ? kb : (Kd - 1);
        float bv = transB ? Bm[(long)nc * ldb + kbc] : Bm[(long)kbc * ldb + nc];
        b[e] = (__bf16)((kbOk && nOk) ? bv : 0.0f);
      }
      acc = __builtin_amdgcn_wmma_f32_16x16x32_bf16(
          false, a, false, b, (short)0, acc, false, false);
    }
    int   n    = n0 + l16;
    int   nc   = (n < N) ? n : (N - 1);
    float badd = (bias1 ? bias1[nc] : 0.0f) + (bias2 ? bias2[nc] : 0.0f);
#pragma unroll
    for (int r = 0; r < 8; ++r) {
      int mm = mT * 16 + r + hlf * 8;
      if (mm < M && n < N) C[(long)mm * ldc + n] = acc[r] + badd;
    }
  }
}

// ---------------------------------------------------------------------------
// Segment softmax prep: per batch, logits = hidden@w5+b5, seg id per position,
// per-segment softmax -> per-position weight w[l] and seg[l].
// ---------------------------------------------------------------------------
__global__ __launch_bounds__(256) void pool_prep(
    const float* __restrict__ hidden, const int* __restrict__ clause,
    const float* __restrict__ w5, const float* __restrict__ b5,
    float* __restrict__ wOut, int* __restrict__ segOut)
{
  __shared__ float sLog[512];
  __shared__ int   sSeg[512];
  __shared__ float sMax[33], sSum[33];
  __shared__ int   sClause[32];
  int b = blockIdx.x, tid = threadIdx.x;
  if (tid < 32) sClause[tid] = clause[b * 32 + tid];
  __syncthreads();
  int wave = tid >> 5, lane = tid & 31;
  float bb5 = b5[0];
  for (int i = 0; i < 64; ++i) {                 // 8 waves * 64 = 512 positions
    int pos = wave * 64 + i;
    const float* row = hidden + (long)(b * 512 + pos) * 768;
    float s = 0.f;
    for (int d = lane; d < 768; d += 32) s += row[d] * w5[d];
    for (int off = 16; off > 0; off >>= 1) s += __shfl_xor(s, off, 32);
    if (lane == 0) sLog[pos] = s + bb5;
  }
  __syncthreads();
  for (int l = tid; l < 512; l += 256) {
    int sg = 0;
#pragma unroll
    for (int c = 0; c < 32; ++c) sg += (sClause[c] <= l) ? 1 : 0;
    sSeg[l] = sg;
  }
  __syncthreads();
  if (tid < 33) {
    float mx = -1e30f;
    for (int l = 0; l < 512; ++l) if (sSeg[l] == tid) mx = fmaxf(mx, sLog[l]);
    float sm = 0.f;
    for (int l = 0; l < 512; ++l) if (sSeg[l] == tid) sm += __expf(sLog[l] - mx);
    sMax[tid] = mx; sSum[tid] = sm;
  }
  __syncthreads();
  for (int l = tid; l < 512; l += 256) {
    int sg = sSeg[l];
    wOut[b * 512 + l]   = __expf(sLog[l] - sMax[sg]) / sSum[sg];
    segOut[b * 512 + l] = sg;
  }
}

// ---------------------------------------------------------------------------
// Streaming weighted segment reduction: P[b,s,d] = sum_{l: seg=s} w[l]*hid[b,l,d]
// grid (B, 3) over 256-wide d-chunks; LDS accumulators, prefetch ahead.
// ---------------------------------------------------------------------------
__global__ __launch_bounds__(256) void pool_accum(
    const float* __restrict__ hidden, const float* __restrict__ w,
    const int* __restrict__ seg, float* __restrict__ P)
{
  __shared__ float acc[33 * 256];
  __shared__ float sw[512];
  __shared__ int   ss[512];
  int b = blockIdx.x, tid = threadIdx.x;
  int c0 = blockIdx.y * 256;
  for (int i = tid; i < 33 * 256; i += 256) acc[i] = 0.f;
  for (int l = tid; l < 512; l += 256) { sw[l] = w[b * 512 + l]; ss[l] = seg[b * 512 + l]; }
  __syncthreads();
  const float* hb = hidden + (long)b * 512 * 768 + c0 + tid;
  for (int l = 0; l < 512; ++l) {
    int lp = (l + 8 < 512) ? (l + 8) : 511;
    __builtin_prefetch(hb + (long)lp * 768, 0, 0);   // global_prefetch_b8
    float v = hb[(long)l * 768];
    acc[ss[l] * 256 + tid] += sw[l] * v;             // thread-exclusive column
  }
  __syncthreads();
  float* Pb = P + (long)b * 33 * 768 + c0 + tid;
  for (int s = 0; s < 33; ++s) Pb[(long)s * 768] = acc[s * 256 + tid];
}

// ---------------------------------------------------------------------------
// Fact path: only segment 0 of segment_pool(hidden_fact) is consumed
// (fact_static_query). Segment 0 = positions l < clause[b,0] (~15..30 rows),
// so we read ~90KB instead of streaming the full 100MB tensor.
// ---------------------------------------------------------------------------
__global__ __launch_bounds__(256) void fact_pool(
    const float* __restrict__ hf, const int* __restrict__ clause,
    const float* __restrict__ w5, const float* __restrict__ b5,
    float* __restrict__ out)
{
  __shared__ float wl[512];
  __shared__ float sMx, sSm;
  int b = blockIdx.x, tid = threadIdx.x;
  int c0 = clause[b * 32];
  if (c0 < 1) c0 = 1; if (c0 > 512) c0 = 512;
  int wave = tid >> 5, lane = tid & 31;
  float bb5 = b5[0];
  for (int pos = wave; pos < c0; pos += 8) {
    const float* row = hf + (long)(b * 512 + pos) * 768;
    float s = 0.f;
    for (int d = lane; d < 768; d += 32) s += row[d] * w5[d];
    for (int off = 16; off > 0; off >>= 1) s += __shfl_xor(s, off, 32);
    if (lane == 0) wl[pos] = s + bb5;
  }
  __syncthreads();
  if (tid == 0) {
    float mx = -1e30f;
    for (int l = 0; l < c0; ++l) mx = fmaxf(mx, wl[l]);
    float sm = 0.f;
    for (int l = 0; l < c0; ++l) sm += __expf(wl[l] - mx);
    sMx = mx; sSm = sm;
  }
  __syncthreads();
  for (int l = tid; l < c0; l += 256) wl[l] = __expf(wl[l] - sMx) / sSm;
  __syncthreads();
  for (int d = tid; d < 768; d += 256) {
    float s = 0.f;
    for (int l = 0; l < c0; ++l) s += wl[l] * hf[(long)(b * 512 + l) * 768 + d];
    out[b * 768 + d] = s;
  }
}

// h2 rows 1..32 = P rows 1..32 (row 0 is written by the Wql GEMM)
__global__ void copy_h2(const float* __restrict__ P, float* __restrict__ h2)
{
  int idx = blockIdx.x * 256 + threadIdx.x;
  if (idx >= 64 * 32 * 768) return;
  int b = idx / (32 * 768);
  int r = idx - b * 32 * 768;
  int off = (b * 33 + 1) * 768 + r;
  h2[off] = P[off];
}

// ---------------------------------------------------------------------------
// LSTM recurrence (xg = x@Wih^T + biases precomputed via WMMA GEMM).
// 8 batches per block, h/c state in LDS, 33 sequential steps.
// ---------------------------------------------------------------------------
__global__ __launch_bounds__(256) void lstm_kernel(
    const float* __restrict__ xg, const float* __restrict__ Whh,
    float* __restrict__ h3, int reverse, int dirOff)
{
  __shared__ float sh[800];
  __shared__ float sc[800];
  int tid = threadIdx.x;
  int bBase = blockIdx.x * 8;
  for (int i = tid; i < 800; i += 256) { sh[i] = 0.f; sc[i] = 0.f; }
  __syncthreads();
  for (int step = 0; step < 33; ++step) {
    int t = reverse ? (32 - step) : step;
    float hn[4], cn[4];
#pragma unroll
    for (int it = 0; it < 4; ++it) {
      int cell = tid + it * 256;
      hn[it] = 0.f; cn[it] = 0.f;
      if (cell < 800) {
        int bb = cell / 100, hi = cell - bb * 100;
        int b = bBase + bb;
        const float* xr = xg + (long)(b * 33 + t) * 400;
        float gi = xr[hi], gf = xr[100 + hi], gg = xr[200 + hi], go = xr[300 + hi];
        const float* wi = Whh + hi * 100;
        const float* wf = Whh + (100 + hi) * 100;
        const float* wg = Whh + (200 + hi) * 100;
        const float* wo = Whh + (300 + hi) * 100;
        const float* hv = sh + bb * 100;
        for (int k = 0; k < 100; ++k) {
          float h = hv[k];
          gi += h * wi[k]; gf += h * wf[k]; gg += h * wg[k]; go += h * wo[k];
        }
        float si = 1.f / (1.f + __expf(-gi));
        float sf = 1.f / (1.f + __expf(-gf));
        float so = 1.f / (1.f + __expf(-go));
        float cc = sf * sc[cell] + si * tanhf(gg);
        hn[it] = so * tanhf(cc);
        cn[it] = cc;
      }
    }
    __syncthreads();
#pragma unroll
    for (int it = 0; it < 4; ++it) {
      int cell = tid + it * 256;
      if (cell < 800) {
        sh[cell] = hn[it]; sc[cell] = cn[it];
        int bb = cell / 100, hi = cell - bb * 100;
        int b = bBase + bb;
        h3[(long)(b * 33 + t) * 200 + dirOff + hi] = hn[it];
      }
    }
    __syncthreads();
  }
}

// ---------------------------------------------------------------------------
// Heads. Attention is analytically degenerate: all K rows equal per batch ->
// softmax uniform -> query_att[b,q,:] = mean(V) = h3[b,0,:]@Wv = vq[b,:].
// h4 = [vq, doc_lstm, |vq - doc_lstm|]; tiny N=3/N=2 heads done in VALU.
// ---------------------------------------------------------------------------
__global__ __launch_bounds__(256) void heads_kernel(
    const float* __restrict__ vq, const float* __restrict__ h3,
    const float* __restrict__ Wp, const float* __restrict__ bp,
    const float* __restrict__ Wc, const float* __restrict__ bc,
    float* __restrict__ outPol, float* __restrict__ outCause)
{
  __shared__ float sWp[1800];
  __shared__ float sWc[1200];
  int tid = threadIdx.x;
  for (int i = tid; i < 1800; i += 256) sWp[i] = Wp[i];
  for (int i = tid; i < 1200; i += 256) sWc[i] = Wc[i];
  __syncthreads();
  int idx = blockIdx.x * 256 + tid;
  if (idx >= 64 * 32) return;
  int b = idx >> 5, q = idx & 31;
  const float* v  = vq + b * 200;
  const float* dl = h3 + (long)(b * 33 + q + 1) * 200;
  float p0 = bp[0], p1 = bp[1], p2 = bp[2];
  float c0 = bc[0], c1 = bc[1];
  for (int f = 0; f < 200; ++f) {
    float xv = v[f], xd = dl[f], xa = fabsf(xv - xd);
    p0 += xv * sWp[f * 3 + 0] + xd * sWp[(200 + f) * 3 + 0] + xa * sWp[(400 + f) * 3 + 0];
    p1 += xv * sWp[f * 3 + 1] + xd * sWp[(200 + f) * 3 + 1] + xa * sWp[(400 + f) * 3 + 1];
    p2 += xv * sWp[f * 3 + 2] + xd * sWp[(200 + f) * 3 + 2] + xa * sWp[(400 + f) * 3 + 2];
    c0 += xv * sWc[f * 2 + 0] + xd * sWc[(200 + f) * 2 + 0] + xa * sWc[(400 + f) * 2 + 0];
    c1 += xv * sWc[f * 2 + 1] + xd * sWc[(200 + f) * 2 + 1] + xa * sWc[(400 + f) * 2 + 1];
  }
  outPol[idx * 3 + 0] = p0; outPol[idx * 3 + 1] = p1; outPol[idx * 3 + 2] = p2;
  outCause[idx * 2 + 0] = c0; outCause[idx * 2 + 1] = c1;
}

// ---------------------------------------------------------------------------
extern "C" void kernel_launch(void* const* d_in, const int* in_sizes, int n_in,
                              void* d_out, int out_size, void* d_ws, size_t ws_size,
                              hipStream_t stream)
{
  const float* hidden_state = (const float*)d_in[0];
  const float* hidden_fact  = (const float*)d_in[1];
  const float* w5    = (const float*)d_in[2];
  const float* b5    = (const float*)d_in[3];
  const float* Wql   = (const float*)d_in[4];
  const float* bql   = (const float*)d_in[5];
  const float* Wfc1  = (const float*)d_in[6];
  const float* bfc1  = (const float*)d_in[7];
  const float* Wih_f = (const float*)d_in[8];
  const float* Whh_f = (const float*)d_in[9];
  const float* bih_f = (const float*)d_in[10];
  const float* bhh_f = (const float*)d_in[11];
  const float* Wih_b = (const float*)d_in[12];
  const float* Whh_b = (const float*)d_in[13];
  const float* bih_b = (const float*)d_in[14];
  const float* bhh_b = (const float*)d_in[15];
  // d_in[16]=Wq, d_in[17]=Wk are mathematically dead (uniform softmax)
  const float* Wv    = (const float*)d_in[18];
  const float* Wc    = (const float*)d_in[19];
  const float* bc    = (const float*)d_in[20];
  const float* Wp    = (const float*)d_in[21];
  const float* bp    = (const float*)d_in[22];
  const float* Wsq   = (const float*)d_in[23];
  const float* bsq   = (const float*)d_in[24];
  const float* Wds   = (const float*)d_in[25];
  const float* bds   = (const float*)d_in[26];
  const float* Wfsq  = (const float*)d_in[27];
  const float* bfsq  = (const float*)d_in[28];
  const int*   clause = (const int*)d_in[29];
  // d_in[30] y_causes unused in forward

  char* ws = (char*)d_ws;
  size_t o = 0;
  auto alloc = [&](size_t bytes) { void* p = ws + o; o = (o + bytes + 255) & ~(size_t)255; return p; };
  float* wBuf = (float*)alloc((size_t)64 * 512 * 4);
  int*   segB = (int*)  alloc((size_t)64 * 512 * 4);
  float* P    = (float*)alloc((size_t)64 * 33 * 768 * 4);
  float* h2   = (float*)alloc((size_t)64 * 33 * 768 * 4);
  float* x    = (float*)alloc((size_t)64 * 33 * 200 * 4);
  float* xgf  = (float*)alloc((size_t)64 * 33 * 400 * 4);
  float* xgb  = (float*)alloc((size_t)64 * 33 * 400 * 4);
  float* h3   = (float*)alloc((size_t)64 * 33 * 200 * 4);
  float* vq   = (float*)alloc((size_t)64 * 200 * 4);
  float* fsq  = (float*)alloc((size_t)64 * 768 * 4);

  float* out      = (float*)d_out;
  float* outPol   = out;            // 6144
  float* outCause = out + 6144;     // 4096
  float* outSq    = out + 10240;    // 49152
  float* outDs1   = out + 59392;    // 1572864
  float* outFsq   = out + 1632256;  // 49152

  auto gemm = [&](const float* A, const float* B, float* C, const float* b1,
                  const float* b2, int M, int N, int K, int lda, int ldb,
                  int ldc, int aGroup, int transB) {
    int tilesN = (N + 15) / 16, tilesM = (M + 15) / 16;
    int total  = tilesN * tilesM;
    gemm_bf16_wmma<<<(total + 7) / 8, 256, 0, stream>>>(
        A, B, C, b1, b2, M, N, K, lda, ldb, ldc, tilesN, total, aGroup, transB);
  };

  // segment pooling (state) + fact segment-0 pooling
  pool_prep <<<64, 256, 0, stream>>>(hidden_state, clause, w5, b5, wBuf, segB);
  pool_accum<<<dim3(64, 3), 256, 0, stream>>>(hidden_state, wBuf, segB, P);
  fact_pool <<<64, 256, 0, stream>>>(hidden_fact, clause, w5, b5, fsq);

  // query = static_query@Wql+bql written straight into h2 row 0 (ldc=33*768)
  gemm(P, Wql, h2, bql, nullptr, 64, 768, 768, 33 * 768, 768, 33 * 768, 0, 0);
  { int total = 64 * 32 * 768;
    copy_h2<<<(total + 255) / 256, 256, 0, stream>>>(P, h2); }

  // x = h2@Wfc1+bfc1
  gemm(h2, Wfc1, x, bfc1, nullptr, 2112, 200, 768, 768, 200, 200, 0, 0);
  // xg = x@Wih^T + (bih+bhh) per direction (transB)
  gemm(x, Wih_f, xgf, bih_f, bhh_f, 2112, 400, 200, 200, 200, 400, 0, 1);
  gemm(x, Wih_b, xgb, bih_b, bhh_b, 2112, 400, 200, 200, 200, 400, 0, 1);

  lstm_kernel<<<8, 256, 0, stream>>>(xgf, Whh_f, h3, 0, 0);
  lstm_kernel<<<8, 256, 0, stream>>>(xgb, Whh_b, h3, 1, 100);

  // vq = h3[:,0,:]@Wv  (only surviving attention GEMM; lda strides batches)
  gemm(h3, Wv, vq, nullptr, nullptr, 64, 200, 200, 33 * 200, 200, 200, 0, 0);
  heads_kernel<<<8, 256, 0, stream>>>(vq, h3, Wp, bp, Wc, bc, outPol, outCause);

  // output projections
  gemm(P,   Wsq,  outSq,  bsq,  nullptr, 64,   768, 768, 33 * 768, 768, 768, 0,  0);
  gemm(P,   Wds,  outDs1, bds,  nullptr, 2048, 768, 768, 768,      768, 768, 32, 0); // row remap m+m/32+1
  gemm(fsq, Wfsq, outFsq, bfsq, nullptr, 64,   768, 768, 768,      768, 768, 0,  0);

  (void)in_sizes; (void)n_in; (void)out_size; (void)ws_size;
}
